// HeatmapDecoder_47519518163425
// MI455X (gfx1250) — matrix operand order repaired and checked by reading, not compiled, not run on hardware
//
#include <hip/hip_runtime.h>
#include <hip/hip_bf16.h>
#include <math.h>

// ---- problem constants (from reference) ----
#define B_   512
#define H_   128
#define T_   60
#define G_   64
#define GG_  4096
#define NS_  6
#define CELL_ 1.5625f      // 100/64
#define HALF_CELL_ 0.78125f

typedef __attribute__((ext_vector_type(16))) _Float16 v16h;
typedef __attribute__((ext_vector_type(8)))  _Float16 v8h;
typedef __attribute__((ext_vector_type(8)))  float    v8f;

// ---------------- helpers ----------------
__device__ __forceinline__ float sigf(float x) { return 1.0f / (1.0f + __expf(-x)); }

__device__ __forceinline__ unsigned pcg(unsigned x) {
    x = x * 747796405u + 2891336453u;
    x = ((x >> ((x >> 28) + 4u)) ^ x) * 277803737u;
    return (x >> 22) ^ x;
}
__device__ __forceinline__ float u01(unsigned h) {
    return (float)(h >> 8) * (1.0f / 16777216.0f) + 1e-7f;
}

// Build v16h A-tile operand from LDS row (16-bit A layout, 16x32 per K-block):
//  element e<8  -> K = kb*32 + 8*hs + e      (contiguous 8 halves, 16B aligned)
//  element e>=8 -> K = kb*32 + 16 + 8*hs + (e-8) (contiguous 8 halves, 16B aligned)
__device__ __forceinline__ v16h load_a16(const _Float16* rowptr, int kb, int hs) {
    v8h a0 = *(const v8h*)(rowptr + kb * 32 + 8 * hs);
    v8h a1 = *(const v8h*)(rowptr + kb * 32 + 16 + 8 * hs);
    v16h A;
#pragma unroll
    for (int e = 0; e < 8; ++e) { A[e] = a0[e]; A[e + 8] = a1[e]; }
    return A;
}

// Build v16h B-tile operand from a global f16 weight row (row-major [n][k], K=128):
//  lane n = nt*16 + (lane&15); element e<8 -> K = kb*32 + 8*hs + e, e>=8 -> K = kb*32 + 16 + 8*hs + (e-8)
__device__ __forceinline__ v16h load_b16(const _Float16* rowptr, int kb, int hs) {
    v8h b0 = *(const v8h*)(rowptr + kb * 32 + 8 * hs);
    v8h b1 = *(const v8h*)(rowptr + kb * 32 + 16 + 8 * hs);
    v16h Bv;
#pragma unroll
    for (int e = 0; e < 8; ++e) { Bv[e] = b0[e]; Bv[e + 8] = b1[e]; }
    return Bv;
}

// ---------------- f32 -> f16 weight conversion ----------------
__global__ void cvt_kernel(const float* __restrict__ src, _Float16* __restrict__ dst, int n) {
    int i = blockIdx.x * 256 + threadIdx.x;
    if (i < n) dst[i] = (_Float16)src[i];
}

// ---------------- fused 2-layer GRU (per-row independent) ----------------
// grid = B_, block = 128. Produces last (B_ x H_) in workspace.
__global__ void gru_kernel(const float* __restrict__ x, const float* __restrict__ hidden,
                           const float* __restrict__ w_ih0, const float* __restrict__ w_hh0,
                           const float* __restrict__ b_ih0, const float* __restrict__ b_hh0,
                           const float* __restrict__ w_ih1, const float* __restrict__ w_hh1,
                           const float* __restrict__ b_ih1, const float* __restrict__ b_hh1,
                           float* __restrict__ last_out) {
    const int b = blockIdx.x, j = threadIdx.x;
    __shared__ float h0s[H_], h1s[H_], x1s[H_];
    h0s[j] = hidden[b * H_ + j];
    h1s[j] = hidden[B_ * H_ + b * H_ + j];
    __syncthreads();

    // ---- layer 0 (x is 2-wide) ----
    const float x0 = x[b * 2 + 0], x1 = x[b * 2 + 1];
    float gir = b_ih0[j]          + x0 * w_ih0[2 * j]              + x1 * w_ih0[2 * j + 1];
    float giz = b_ih0[H_ + j]     + x0 * w_ih0[2 * (H_ + j)]       + x1 * w_ih0[2 * (H_ + j) + 1];
    float gin = b_ih0[2 * H_ + j] + x0 * w_ih0[2 * (2 * H_ + j)]   + x1 * w_ih0[2 * (2 * H_ + j) + 1];
    float ghr = b_hh0[j], ghz = b_hh0[H_ + j], ghn = b_hh0[2 * H_ + j];
#pragma unroll 4
    for (int k = 0; k < H_; ++k) {
        const float h = h0s[k];
        ghr += h * w_hh0[j * H_ + k];
        ghz += h * w_hh0[(H_ + j) * H_ + k];
        ghn += h * w_hh0[(2 * H_ + j) * H_ + k];
    }
    float r = sigf(gir + ghr), z = sigf(giz + ghz);
    float n = tanhf(gin + r * ghn);
    x1s[j] = (1.0f - z) * n + z * h0s[j];     // h0 -> input of layer 1
    __syncthreads();

    // ---- layer 1 (K=128 both paths) ----
    float gir1 = b_ih1[j], giz1 = b_ih1[H_ + j], gin1 = b_ih1[2 * H_ + j];
    float ghr1 = b_hh1[j], ghz1 = b_hh1[H_ + j], ghn1 = b_hh1[2 * H_ + j];
#pragma unroll 4
    for (int k = 0; k < H_; ++k) {
        const float xv = x1s[k], hv = h1s[k];
        gir1 += xv * w_ih1[j * H_ + k];
        giz1 += xv * w_ih1[(H_ + j) * H_ + k];
        gin1 += xv * w_ih1[(2 * H_ + j) * H_ + k];
        ghr1 += hv * w_hh1[j * H_ + k];
        ghz1 += hv * w_hh1[(H_ + j) * H_ + k];
        ghn1 += hv * w_hh1[(2 * H_ + j) * H_ + k];
    }
    float r1 = sigf(gir1 + ghr1), z1 = sigf(giz1 + ghz1);
    float n1 = tanhf(gin1 + r1 * ghn1);
    last_out[b * H_ + j] = (1.0f - z1) * n1 + z1 * h1s[j];
}

// ---------------- trajectory + confidence heads ----------------
// grid = B_, block = 128. Writes predictions[:,0,:,:] and confidences.
__global__ void heads_kernel(const float* __restrict__ last,
                             const float* __restrict__ hg_w1, const float* __restrict__ hg_b1,
                             const float* __restrict__ hg_w2, const float* __restrict__ hg_b2,
                             const float* __restrict__ hg_w3, const float* __restrict__ hg_b3,
                             const float* __restrict__ cf_w1, const float* __restrict__ cf_b1,
                             const float* __restrict__ cf_w2, const float* __restrict__ cf_b2,
                             float* __restrict__ pred_out, float* __restrict__ conf_out) {
    const int b = blockIdx.x, j = threadIdx.x;
    __shared__ float ls[H_], t1[H_], t2[64], c1[64], cv[64];
    ls[j] = last[b * H_ + j];
    __syncthreads();

    {
        float a = hg_b1[j];
#pragma unroll 4
        for (int k = 0; k < H_; ++k) a += ls[k] * hg_w1[j * H_ + k];
        t1[j] = a > 0.0f ? a : 0.0f;
    }
    if (j < 64) {
        float a = cf_b1[j];
#pragma unroll 4
        for (int k = 0; k < H_; ++k) a += ls[k] * cf_w1[j * H_ + k];
        c1[j] = a > 0.0f ? a : 0.0f;
    }
    __syncthreads();

    if (j < 64) {
        float a = hg_b2[j];
#pragma unroll 4
        for (int k = 0; k < H_; ++k) a += t1[k] * hg_w2[j * H_ + k];
        t2[j] = a > 0.0f ? a : 0.0f;
    }
    if (j < T_) {
        float a = cf_b2[j];
#pragma unroll 4
        for (int k = 0; k < 64; ++k) a += c1[k] * cf_w2[j * 64 + k];
        cv[j] = a;
    }
    __syncthreads();

    if (j < 2 * T_) {  // traj: output dim 120 -> (T,2)
        float a = hg_b3[j];
#pragma unroll 4
        for (int k = 0; k < 64; ++k) a += t2[k] * hg_w3[j * 64 + k];
        const int tt = j >> 1, c = j & 1;
        pred_out[(((long)b * NS_ + 0) * T_ + tt) * 2 + c] = a;
    }
    if (j == 0) {
        float s = 0.0f;
        for (int i = 0; i < T_; ++i) s += cv[i];
        const float m = s * (1.0f / (float)T_);
        float d = 1.0f;
        for (int sm = 0; sm < NS_; ++sm) { conf_out[b * NS_ + sm] = m * d; d *= 0.9f; }
    }
}

// ---------------- fused noisy -> hph -> probs -> inverse-CDF sampling ----------------
// grid = (B_/16, T_), block = 256 (8 waves). Dynamic LDS:
//   probs : 16 x 4096 f32 (256 KB)   hph : 16 x 128 f16   noisy : 16 x 128 f16
__global__ void __launch_bounds__(256)
heatmap_sample_kernel(const float* __restrict__ last,
                      const _Float16* __restrict__ w1h, const float* __restrict__ b1,
                      const _Float16* __restrict__ w2h, const float* __restrict__ b2,
                      float* __restrict__ pred_out) {
    extern __shared__ char smem_raw[];
    float*    probs = (float*)smem_raw;                          // 16*GG_ floats (exp(logit))
    _Float16* hph   = (_Float16*)(smem_raw + 16 * GG_ * 4);      // 16*H_ halves
    _Float16* noisy = hph + 16 * H_;                             // 16*H_ halves
    __shared__ float rowsum[16];                                 // per-row softmax denominators

    const int tid  = threadIdx.x;
    const int lane = tid & 31;
    const int wv   = tid >> 5;
    const int hs   = lane >> 4;      // half-wave select
    const int ln   = lane & 15;
    const int b0   = blockIdx.x * 16;
    const int t    = blockIdx.y;
    const float scale_t = 0.1f * (float)t / (float)T_;

    if (tid < 16) rowsum[tid] = 0.0f;

    // (1) noisy = last + N(0,1)*scale_t, in f16 (16 x 128)
    for (int i = tid; i < 16 * H_; i += 256) {
        const int m = i >> 7, k = i & (H_ - 1);
        unsigned seed = (unsigned)((t * B_ + (b0 + m)) * H_ + k) * 2654435761u + 12345u;
        unsigned h1 = pcg(seed), h2 = pcg(h1 ^ 0x9e3779b9u);
        const float g = sqrtf(-2.0f * __logf(u01(h1))) * __cosf(6.28318530718f * u01(h2));
        noisy[i] = (_Float16)(last[(b0 + m) * H_ + k] + scale_t * g);
    }
    __syncthreads();

    // (2) hph = relu(noisy @ hp_w1^T + b1): 16x128, one 16-col N-tile per wave
    {
        const int nt = wv;                       // 8 waves == 8 N-tiles
        v8f acc = {};
#pragma unroll
        for (int kb = 0; kb < 4; ++kb) {
            v16h A  = load_a16(noisy + ln * H_, kb, hs);
            v16h Bv = load_b16(w1h + (nt * 16 + ln) * H_, kb, hs);
            acc = __builtin_amdgcn_wmma_f32_16x16x32_f16(false, A, false, Bv,
                                                         (short)0, acc, false, false);
        }
        const float bias = b1[nt * 16 + ln];
#pragma unroll
        for (int r = 0; r < 8; ++r) {
            float v = acc[r] + bias;
            v = v > 0.0f ? v : 0.0f;
            hph[(r + 8 * hs) * H_ + nt * 16 + ln] = (_Float16)v;
        }
    }
    __syncthreads();

    // (3) probs = exp(hph @ hp_w2^T + b2): 16x4096, 32 N-tiles per wave,
    //     double-buffered B loads; per-row partial sums in registers.
    {
        v16h Areg[4];
#pragma unroll
        for (int kb = 0; kb < 4; ++kb) Areg[kb] = load_a16(hph + ln * H_, kb, hs);

        float psum[8];
#pragma unroll
        for (int r = 0; r < 8; ++r) psum[r] = 0.0f;

        const int nt0 = wv * 32;
        v16h Bp[4], Bq[4];
#pragma unroll
        for (int kb = 0; kb < 4; ++kb)
            Bp[kb] = load_b16(w2h + (nt0 * 16 + ln) * H_, kb, hs);

        for (int nti = 0; nti < 32; nti += 2) {
            // prefetch tile nti+1 while computing nti
#pragma unroll
            for (int kb = 0; kb < 4; ++kb)
                Bq[kb] = load_b16(w2h + ((nt0 + nti + 1) * 16 + ln) * H_, kb, hs);
            {
                const int nt = nt0 + nti;
                v8f acc = {};
#pragma unroll
                for (int kb = 0; kb < 4; ++kb)
                    acc = __builtin_amdgcn_wmma_f32_16x16x32_f16(false, Areg[kb], false, Bp[kb],
                                                                 (short)0, acc, false, false);
                const float bias = b2[nt * 16 + ln];
#pragma unroll
                for (int r = 0; r < 8; ++r) {
                    float lv = acc[r] + bias;
                    lv = fminf(fmaxf(lv, -60.0f), 60.0f);   // overflow guard (scale-invariant)
                    const float p = __expf(lv);
                    psum[r] += p;
                    probs[(r + 8 * hs) * GG_ + nt * 16 + ln] = p;
                }
            }
            // prefetch tile nti+2 while computing nti+1
            if (nti + 2 < 32) {
#pragma unroll
                for (int kb = 0; kb < 4; ++kb)
                    Bp[kb] = load_b16(w2h + ((nt0 + nti + 2) * 16 + ln) * H_, kb, hs);
            }
            {
                const int nt = nt0 + nti + 1;
                v8f acc = {};
#pragma unroll
                for (int kb = 0; kb < 4; ++kb)
                    acc = __builtin_amdgcn_wmma_f32_16x16x32_f16(false, Areg[kb], false, Bq[kb],
                                                                 (short)0, acc, false, false);
                const float bias = b2[nt * 16 + ln];
#pragma unroll
                for (int r = 0; r < 8; ++r) {
                    float lv = acc[r] + bias;
                    lv = fminf(fmaxf(lv, -60.0f), 60.0f);
                    const float p = __expf(lv);
                    psum[r] += p;
                    probs[(r + 8 * hs) * GG_ + nt * 16 + ln] = p;
                }
            }
        }

        // reduce psum across the 16 lanes of each half-wave, then one LDS atomic per row
#pragma unroll
        for (int r = 0; r < 8; ++r) {
#pragma unroll
            for (int off = 1; off < 16; off <<= 1)
                psum[r] += __shfl_xor(psum[r], off, 32);
        }
        if (ln == 0) {
#pragma unroll
            for (int r = 0; r < 8; ++r)
                atomicAdd(&rowsum[r + 8 * hs], psum[r]);
        }
    }
    __syncthreads();

    // (4) inverse-CDF categorical sampling: 5 samples x 16 rows = 80 wave-tasks.
    //     Wave walks the row's prob array with an inclusive prefix scan, early-out.
    for (int task = wv; task < 5 * 16; task += 8) {
        const int s = task >> 4;     // 0..4 -> output sample s+1
        const int r = task & 15;
        const float* row = probs + r * GG_;

        const unsigned h = pcg(0x12345u ^ (unsigned)(((s * T_ + t) * B_ + (b0 + r)) * 2654435761u));
        const float target = u01(h) * rowsum[r];

        int   idx = GG_ - 1;
        float running = 0.0f;
        for (int c = 0; c < GG_ / 32; ++c) {
            float xv = row[c * 32 + lane];
#pragma unroll
            for (int off = 1; off < 32; off <<= 1) {
                const float y = __shfl_up(xv, off, 32);
                if (lane >= off) xv += y;
            }
            const float tot = __shfl(xv, 31, 32);
            const unsigned long long m = __ballot(running + xv >= target);
            if (m) { idx = c * 32 + (__ffsll(m) - 1); break; }
            running += tot;
        }

        if (lane == 0) {
            const int xi = idx & (G_ - 1), yi = idx >> 6;
            const float px = -50.0f + (float)xi * CELL_ + HALF_CELL_;
            const float py = -50.0f + (float)yi * CELL_ + HALF_CELL_;
            const long o = (((long)(b0 + r) * NS_ + (s + 1)) * T_ + t) * 2;
            pred_out[o + 0] = px;
            pred_out[o + 1] = py;
        }
    }
}

// ---------------- host launcher ----------------
extern "C" void kernel_launch(void* const* d_in, const int* in_sizes, int n_in,
                              void* d_out, int out_size, void* d_ws, size_t ws_size,
                              hipStream_t stream) {
    (void)in_sizes; (void)n_in; (void)out_size; (void)ws_size;
    const float* x       = (const float*)d_in[0];
    const float* hidden  = (const float*)d_in[1];
    // d_in[2] = num_samples (fixed 6 per reference setup)
    const float* w_ih_l0 = (const float*)d_in[3];
    const float* w_hh_l0 = (const float*)d_in[4];
    const float* b_ih_l0 = (const float*)d_in[5];
    const float* b_hh_l0 = (const float*)d_in[6];
    const float* w_ih_l1 = (const float*)d_in[7];
    const float* w_hh_l1 = (const float*)d_in[8];
    const float* b_ih_l1 = (const float*)d_in[9];
    const float* b_hh_l1 = (const float*)d_in[10];
    const float* hg_w1   = (const float*)d_in[11];
    const float* hg_b1   = (const float*)d_in[12];
    const float* hg_w2   = (const float*)d_in[13];
    const float* hg_b2   = (const float*)d_in[14];
    const float* hg_w3   = (const float*)d_in[15];
    const float* hg_b3   = (const float*)d_in[16];
    const float* cf_w1   = (const float*)d_in[17];
    const float* cf_b1   = (const float*)d_in[18];
    const float* cf_w2   = (const float*)d_in[19];
    const float* cf_b2   = (const float*)d_in[20];
    const float* hp_w1   = (const float*)d_in[21];
    const float* hp_b1   = (const float*)d_in[22];
    const float* hp_w2   = (const float*)d_in[23];
    const float* hp_b2   = (const float*)d_in[24];

    // workspace layout
    float*    last = (float*)d_ws;                                  // 512*128 f32 (256 KB)
    _Float16* w1h  = (_Float16*)((char*)d_ws + 262144);             // 128*128 f16 (32 KB)
    _Float16* w2h  = (_Float16*)((char*)d_ws + 262144 + 32768);     // 4096*128 f16 (1 MB)

    float* pred     = (float*)d_out;                // B*NS*T*2
    float* conf_out = pred + (long)B_ * NS_ * T_ * 2;

    cvt_kernel<<<(H_ * H_ + 255) / 256, 256, 0, stream>>>(hp_w1, w1h, H_ * H_);
    cvt_kernel<<<(GG_ * H_ + 255) / 256, 256, 0, stream>>>(hp_w2, w2h, GG_ * H_);

    gru_kernel<<<B_, H_, 0, stream>>>(x, hidden,
                                      w_ih_l0, w_hh_l0, b_ih_l0, b_hh_l0,
                                      w_ih_l1, w_hh_l1, b_ih_l1, b_hh_l1, last);

    heads_kernel<<<B_, H_, 0, stream>>>(last,
                                        hg_w1, hg_b1, hg_w2, hg_b2, hg_w3, hg_b3,
                                        cf_w1, cf_b1, cf_w2, cf_b2,
                                        pred, conf_out);

    const size_t lds_bytes = (size_t)16 * GG_ * 4 + 2 * (size_t)16 * H_ * 2;  // 270336
    dim3 grid(B_ / 16, T_);
    heatmap_sample_kernel<<<grid, 256, lds_bytes, stream>>>(last, w1h, hp_b1, w2h, hp_b2, pred);
}
